// RQ_34720515621660
// MI455X (gfx1250) — matrix operand order repaired and checked by reading, not compiled, not run on hardware
//
#include <hip/hip_runtime.h>
#include <hip/hip_bf16.h>

#define N_ROWS 32768
#define DIM    768
#define NQ     4
#define NV     4096
#define LDSB_STRIDE 776   // 768 + 8 bf16 pad to break LDS bank alignment

typedef __attribute__((ext_vector_type(16))) __bf16 v16bf;
typedef __attribute__((ext_vector_type(8)))  __bf16 v8bf;
typedef __attribute__((ext_vector_type(8)))  float  v8f;
typedef __attribute__((ext_vector_type(4)))  int    v4i;

#if defined(__gfx1250__) && __has_builtin(__builtin_amdgcn_global_load_async_to_lds_b128)
#define RQ_ASYNC_LDS 1
typedef __attribute__((address_space(1))) v4i glb_v4i;   // global (prints as __device__)
typedef __attribute__((address_space(3))) v4i lds_v4i;   // LDS (prints as __shared__)
#else
#define RQ_ASYNC_LDS 0
#endif

// ---------- prep: bf16 codebook copy + ||c||^2 per codeword ----------
__global__ void rq_prep_cb(const float* __restrict__ cb,
                           __hip_bfloat16* __restrict__ cbb,
                           float* __restrict__ cnorm) {
  const int b = blockIdx.x;                   // one of Q*V codewords
  const size_t base = (size_t)b * DIM;
  float ss = 0.f;
  for (int j = threadIdx.x; j < DIM; j += blockDim.x) {
    float v = cb[base + j];
    cbb[base + j] = __float2bfloat16(v);
    ss += v * v;
  }
  for (int off = 16; off > 0; off >>= 1) ss += __shfl_down(ss, off, 32);
  __shared__ float parts[8];
  const int wid = threadIdx.x >> 5, lane = threadIdx.x & 31;
  if (lane == 0) parts[wid] = ss;
  __syncthreads();
  if (threadIdx.x == 0) {
    float t = 0.f;
    for (int i = 0; i < (int)(blockDim.x >> 5); ++i) t += parts[i];
    cnorm[b] = t;
  }
}

// ---------- init residual (f32 exact + bf16 mirror for WMMA) ----------
__global__ void rq_init_res(const float* __restrict__ x,
                            float* __restrict__ rf,
                            __hip_bfloat16* __restrict__ rb, long total) {
  long i = (long)blockIdx.x * blockDim.x + threadIdx.x;
  const long stride = (long)gridDim.x * blockDim.x;
  for (; i < total; i += stride) {
    float v = x[i];
    rf[i] = v;
    rb[i] = __float2bfloat16(v);
  }
}

// ---------- score GEMM + row argmax for one level ----------
// Block: 128 threads (4 waves), 64 rows. Each wave = 16-row stripe.
// Per 32-codeword chunk: stage B into LDS (async global->LDS when available,
// shared by all 4 waves), then 24 k-steps x 2 WMMA per wave.
__global__ void __launch_bounds__(128)
rq_argmax(const __hip_bfloat16* __restrict__ resb,
          const __hip_bfloat16* __restrict__ cbb,    // this level: V*D bf16
          const float* __restrict__ cnorm,           // this level: V
          int* __restrict__ idx_out) {
  __shared__ __bf16 Bsh[32 * LDSB_STRIDE];

  const int tid  = threadIdx.x;
  const int wave = tid >> 5;
  const int lane = tid & 31;
  const int half = lane >> 4;      // 0: lanes 0-15, 1: lanes 16-31
  const int l16  = lane & 15;

  const int rowbase = blockIdx.x * 64 + wave * 16;
  // A-fragment source row for this lane (ISA 16-bit A layout: M = lane%16)
  const __bf16* arow = (const __bf16*)resb + (size_t)(rowbase + l16) * DIM;
  const __bf16* cbp  = (const __bf16*)cbb;

  float rmax[8];
  int   ridx[8];
#pragma unroll
  for (int i = 0; i < 8; ++i) { rmax[i] = -3.4e38f; ridx[i] = 0; }

  for (int chunk = 0; chunk < NV / 32; ++chunk) {
    const int cbase = chunk * 32;
    __syncthreads();                       // protect Bsh reuse
    // stage 32 codewords (32 x 768 bf16) into LDS, 16B granules
#if RQ_ASYNC_LDS
    for (int i = tid; i < 32 * (DIM / 8); i += 128) {
      const int col = i / (DIM / 8);
      const int oct = i - col * (DIM / 8);
      __builtin_amdgcn_global_load_async_to_lds_b128(
          (glb_v4i*)(cbp + (size_t)(cbase + col) * DIM + oct * 8),
          (lds_v4i*)(&Bsh[col * LDSB_STRIDE + oct * 8]),
          0, 0);
    }
#if __has_builtin(__builtin_amdgcn_s_wait_asynccnt)
    __builtin_amdgcn_s_wait_asynccnt(0);
#else
    asm volatile("s_wait_asynccnt 0" ::: "memory");
#endif
#else
    for (int i = tid; i < 32 * (DIM / 8); i += 128) {
      const int col = i / (DIM / 8);
      const int oct = i - col * (DIM / 8);
      v8bf d = *(const v8bf*)(cbp + (size_t)(cbase + col) * DIM + oct * 8);
      *(v8bf*)(&Bsh[col * LDSB_STRIDE + oct * 8]) = d;
    }
#endif
    __syncthreads();

    v8f c0 = {};
    v8f c1 = {};
#pragma unroll 4
    for (int kt = 0; kt < DIM / 32; ++kt) {
      // A 16x32 bf16 fragment: lanes<16 hold K {0..15}, lanes>=16 K {16..31}
      union { v16bf v; v8bf h[2]; } a;
      const int ak = kt * 32 + half * 16;
      a.h[0] = *(const v8bf*)(arow + ak);
      a.h[1] = *(const v8bf*)(arow + ak + 8);
      // B 32x16 bf16 fragments: lane = N, 16 contiguous K per half
      union { v16bf v; v8bf h[2]; } b0, b1;
      const int bk = kt * 32 + half * 16;
      const __bf16* bp0 = &Bsh[l16 * LDSB_STRIDE + bk];
      const __bf16* bp1 = &Bsh[(l16 + 16) * LDSB_STRIDE + bk];
      b0.h[0] = *(const v8bf*)(bp0);
      b0.h[1] = *(const v8bf*)(bp0 + 8);
      b1.h[0] = *(const v8bf*)(bp1);
      b1.h[1] = *(const v8bf*)(bp1 + 8);
      c0 = __builtin_amdgcn_wmma_f32_16x16x32_bf16(false, a.v, false, b0.v,
                                                   (short)0, c0, false, false);
      c1 = __builtin_amdgcn_wmma_f32_16x16x32_bf16(false, a.v, false, b1.v,
                                                   (short)0, c1, false, false);
    }

    // scores = 2*dot - ||c||^2 ; update running argmax (strict > keeps first max)
    const float cn0 = cnorm[cbase + l16];
    const float cn1 = cnorm[cbase + 16 + l16];
#pragma unroll
    for (int i = 0; i < 8; ++i) {
      const float s0 = 2.f * c0[i] - cn0;
      if (s0 > rmax[i]) { rmax[i] = s0; ridx[i] = cbase + l16; }
      const float s1 = 2.f * c1[i] - cn1;
      if (s1 > rmax[i]) { rmax[i] = s1; ridx[i] = cbase + 16 + l16; }
    }
  }

  // reduce across the 16 lanes sharing each row (lower index wins ties)
#pragma unroll
  for (int off = 8; off > 0; off >>= 1) {
#pragma unroll
    for (int i = 0; i < 8; ++i) {
      const float om = __shfl_xor(rmax[i], off, 16);
      const int   oi = __shfl_xor(ridx[i], off, 16);
      if (om > rmax[i] || (om == rmax[i] && oi < ridx[i])) {
        rmax[i] = om; ridx[i] = oi;
      }
    }
  }
  if (l16 == 0) {
#pragma unroll
    for (int i = 0; i < 8; ++i)
      idx_out[rowbase + half * 8 + i] = ridx[i];   // C layout: M = i + 8*half
  }
}

// ---------- gather + residual/output update (exact fp32) ----------
__global__ void __launch_bounds__(192)
rq_update(const float* __restrict__ cbq,       // this level, f32 V*D
          const int* __restrict__ idx,
          float* __restrict__ rf,
          __hip_bfloat16* __restrict__ rb,
          float* __restrict__ out,
          int first) {
  const int row = blockIdx.x;
  const int id  = idx[row];
  const size_t rbase = (size_t)row * DIM;
  const size_t cbase = (size_t)id * DIM;
  const int j = threadIdx.x * 4;               // 192 threads * 4 = 768

  float4 cv = *(const float4*)(cbq + cbase + j);
  float4 rv = *(const float4*)(rf + rbase + j);
  rv.x -= cv.x; rv.y -= cv.y; rv.z -= cv.z; rv.w -= cv.w;
  *(float4*)(rf + rbase + j) = rv;

  __hip_bfloat162* rb2 = (__hip_bfloat162*)(rb + rbase + j);
  __hip_bfloat162 p0, p1;
  p0.x = __float2bfloat16(rv.x); p0.y = __float2bfloat16(rv.y);
  p1.x = __float2bfloat16(rv.z); p1.y = __float2bfloat16(rv.w);
  rb2[0] = p0; rb2[1] = p1;

  float4 ov;
  if (first) {
    ov = cv;
  } else {
    ov = *(const float4*)(out + rbase + j);
    ov.x += cv.x; ov.y += cv.y; ov.z += cv.z; ov.w += cv.w;
  }
  *(float4*)(out + rbase + j) = ov;
}

// ---------- indices -> tail of d_out (float, [N,Q]) ----------
__global__ void rq_write_idx(const int* __restrict__ idx,    // [Q][N]
                             float* __restrict__ outi) {     // [N][Q]
  const int t = blockIdx.x * blockDim.x + threadIdx.x;
  if (t >= N_ROWS * NQ) return;
  const int n = t / NQ, q = t - n * NQ;
  outi[t] = (float)idx[q * N_ROWS + n];
}

extern "C" void kernel_launch(void* const* d_in, const int* in_sizes, int n_in,
                              void* d_out, int out_size, void* d_ws, size_t ws_size,
                              hipStream_t stream) {
  (void)in_sizes; (void)n_in; (void)out_size; (void)ws_size;
  const float* x  = (const float*)d_in[0];
  const float* cb = (const float*)d_in[1];

  char* ws = (char*)d_ws;
  size_t off = 0;
  __hip_bfloat16* cbb = (__hip_bfloat16*)(ws + off); off += (size_t)NQ * NV * DIM * 2;
  float* cnorm        = (float*)(ws + off);          off += (size_t)NQ * NV * 4;
  float* rf           = (float*)(ws + off);          off += (size_t)N_ROWS * DIM * 4;
  __hip_bfloat16* rb  = (__hip_bfloat16*)(ws + off); off += (size_t)N_ROWS * DIM * 2;
  int* idx            = (int*)(ws + off);            off += (size_t)NQ * N_ROWS * 4;
  float* out = (float*)d_out;

  rq_prep_cb<<<NQ * NV, 256, 0, stream>>>(cb, cbb, cnorm);
  rq_init_res<<<2048, 256, 0, stream>>>(x, rf, rb, (long)N_ROWS * DIM);

  for (int q = 0; q < NQ; ++q) {
    rq_argmax<<<N_ROWS / 64, 128, 0, stream>>>(
        rb, cbb + (size_t)q * NV * DIM, cnorm + (size_t)q * NV,
        idx + (size_t)q * N_ROWS);
    rq_update<<<N_ROWS, 192, 0, stream>>>(
        cb + (size_t)q * NV * DIM, idx + (size_t)q * N_ROWS,
        rf, rb, out, q == 0);
  }
  rq_write_idx<<<(N_ROWS * NQ + 255) / 256, 256, 0, stream>>>(
      idx, out + (size_t)N_ROWS * DIM);
}
